// CenterFormerBboxHead_24773371363576
// MI455X (gfx1250) — compile-verified, matrix operand order checked by reading, not run on hardware
//
#include <hip/hip_runtime.h>
#include <hip/hip_bf16.h>

typedef __attribute__((ext_vector_type(16))) _Float16 v16h;
typedef __attribute__((ext_vector_type(8)))  _Float16 v8h;
typedef __attribute__((ext_vector_type(8)))  float    v8f;

#define CIN  256
#define CS   64
#define NPTS 16384
#define NB   8
#define HN   6
#define BN_EPS 1e-3f

// d_ws layout (element = _Float16 unless noted):
//   WsE  [64][256]      halfs [0      .. 16384)
//   W1E  [6][64][64]    halfs [16384  .. 40960)
//   W2E  [6][16][64]    halfs [40960  .. 47104)   (rows >=3 zero-padded)
//   then f32: bsE[64], b1E[6*64], b2E[6*16]
#define WSE_H  0
#define W1E_H  16384
#define W2E_H  40960
#define WTOT_H 47104
#define BIAS_OFF_BYTES (WTOT_H * 2)

// ---------------------------------------------------------------------------
// Prep: fold BN into weights, convert to f16, pad W2 to 16 out rows.
// ---------------------------------------------------------------------------
__global__ __launch_bounds__(256) void prep_kernel(
    const float* __restrict__ sw, const float* __restrict__ sb,
    const float* __restrict__ sg, const float* __restrict__ sbe,
    const float* __restrict__ sm, const float* __restrict__ sv,
    const float* __restrict__ w1, const float* __restrict__ b1,
    const float* __restrict__ g1, const float* __restrict__ be1,
    const float* __restrict__ m1, const float* __restrict__ v1,
    const float* __restrict__ w2, const float* __restrict__ b2,
    _Float16* __restrict__ wsE, _Float16* __restrict__ w1E,
    _Float16* __restrict__ w2E, float* __restrict__ bsE,
    float* __restrict__ b1E, float* __restrict__ b2E)
{
    int i = blockIdx.x * 256 + threadIdx.x;

    if (i < CS * CIN) {                       // shared conv weights
        int o = i >> 8;
        float inv = sg[o] * rsqrtf(sv[o] + BN_EPS);
        wsE[i] = (_Float16)(sw[i] * inv);
        if ((i & 255) == 0)
            bsE[o] = sb[o] * inv + sbe[o] - sm[o] * inv;
    }
    if (i < HN * CS * CS) {                   // head first convs
        int ho = i >> 6;
        float inv = g1[ho] * rsqrtf(v1[ho] + BN_EPS);
        w1E[i] = (_Float16)(w1[i] * inv);
        if ((i & 63) == 0)
            b1E[ho] = b1[ho] * inv + be1[ho] - m1[ho] * inv;
    }
    if (i < HN * 16 * CS) {                   // head final convs, padded 3->16
        int c = i & 63;
        int o = (i >> 6) & 15;
        int h = i >> 10;
        float v = (o < 3) ? w2[(h * 3 + o) * CS + c] : 0.0f;
        w2E[i] = (_Float16)v;
        if (c == 0)
            b2E[h * 16 + o] = (o < 3) ? b2[h * 3 + o] : 0.0f;
    }
}

// ---------------------------------------------------------------------------
// Fused chain: x -> shared conv -> 6 heads, one 16-column N-tile per wave.
// ---------------------------------------------------------------------------
__global__ __launch_bounds__(256) void fused_head_kernel(
    const float* __restrict__ x, const _Float16* __restrict__ wts,
    const float* __restrict__ bsE, const float* __restrict__ b1E,
    const float* __restrict__ b2E, float* __restrict__ out)
{
    __shared__ __align__(32) _Float16 lds_w[WTOT_H];           // 94208 B
    __shared__ __align__(32) _Float16 lds_yt[2 * 8 * 16 * 64]; // 32768 B

    // Cooperative weight stage: global (f16, prepped) -> LDS, 16B vectors.
    {
        const uint4* src = (const uint4*)wts;
        uint4*       dst = (uint4*)lds_w;
        for (int i = threadIdx.x; i < WTOT_H / 8; i += 256)
            dst[i] = src[i];
    }
    __syncthreads();

    const int w    = threadIdx.x >> 5;
    const int lane = threadIdx.x & 31;
    const int col  = lane & 15;   // N within 16-wide tile / A row within 16
    const int hi   = lane >> 4;   // K-half selector for A/B, M+8 half for C
    const int b    = blockIdx.x >> 7;
    const int n0   = (blockIdx.x & 127) * 128 + w * 16;

    const float* xb = x + (size_t)b * CIN * NPTS + n0 + col;

    // ---- Stage 1: y[64x16] = relu(WsE[64x256] @ x[256x16] + bsE) ----------
    v8f acc[4] = {};
#pragma unroll
    for (int kk = 0; kk < 8; ++kk) {
        const int cbase = kk * 32 + hi * 16;
        if (kk < 7)  // prefetch next K-chunk of the only HBM-heavy stream
            __builtin_prefetch(xb + (size_t)(cbase + 32) * NPTS, 0, 0);
        // B fragment: lane holds column `col`, K = kk*32 + hi*16 + j
        v16h bfrag;
#pragma unroll
        for (int j = 0; j < 16; ++j)
            bfrag[j] = (_Float16)xb[(size_t)(cbase + j) * NPTS];
#pragma unroll
        for (int mb = 0; mb < 4; ++mb) {
            const int row = mb * 16 + col;
            v16h afrag = *(const v16h*)&lds_w[WSE_H + row * CIN + cbase];
            acc[mb] = __builtin_amdgcn_wmma_f32_16x16x32_f16(
                false, afrag, false, bfrag, (short)0, acc[mb], false, false);
        }
    }

    // Bias + ReLU, store y to per-wave LDS scratch in B layout: [col][k]
    _Float16* ylds = &lds_yt[w * 1024];
#pragma unroll
    for (int mb = 0; mb < 4; ++mb) {
        const int m0 = mb * 16 + hi * 8;   // C layout: lane holds rows m0..m0+7
        v8h tmp;
#pragma unroll
        for (int r = 0; r < 8; ++r) {
            float v = acc[mb][r] + bsE[m0 + r];
            tmp[r] = (_Float16)fmaxf(v, 0.0f);
        }
        *(v8h*)&ylds[col * CS + m0] = tmp;
    }
    asm volatile("s_wait_dscnt 0" ::: "memory");

    // ---- Stage 2+3: per head (fully unrolled: channel counts fold) --------
    constexpr int kChOff[HN] = {0, 3, 5, 6, 9, 11};
    constexpr int kHCh[HN]   = {3, 2, 1, 3, 2, 1};

    _Float16* tlds = &lds_yt[8192 + w * 1024];
#pragma unroll
    for (int h = 0; h < HN; ++h) {
        const _Float16* w1h = &lds_w[W1E_H + h * CS * CS];   // [64][64]
        v8f a2[4] = {};
#pragma unroll
        for (int kk = 0; kk < 2; ++kk) {
            const int kb = kk * 32 + hi * 16;
            v16h bfrag = *(const v16h*)&ylds[col * CS + kb];
#pragma unroll
            for (int mb = 0; mb < 4; ++mb) {
                v16h afrag = *(const v16h*)&w1h[(mb * 16 + col) * CS + kb];
                a2[mb] = __builtin_amdgcn_wmma_f32_16x16x32_f16(
                    false, afrag, false, bfrag, (short)0, a2[mb], false, false);
            }
        }
#pragma unroll
        for (int mb = 0; mb < 4; ++mb) {
            const int m0 = mb * 16 + hi * 8;
            v8h tmp;
#pragma unroll
            for (int r = 0; r < 8; ++r) {
                float v = a2[mb][r] + b1E[h * CS + m0 + r];
                tmp[r] = (_Float16)fmaxf(v, 0.0f);
            }
            *(v8h*)&tlds[col * CS + m0] = tmp;
        }
        asm volatile("s_wait_dscnt 0" ::: "memory");

        const _Float16* w2h = &lds_w[W2E_H + h * 16 * CS];   // [16][64]
        v8f a3 = {};
#pragma unroll
        for (int kk = 0; kk < 2; ++kk) {
            const int kb = kk * 32 + hi * 16;
            v16h bfrag = *(const v16h*)&tlds[col * CS + kb];
            v16h afrag = *(const v16h*)&w2h[col * CS + kb];
            a3 = __builtin_amdgcn_wmma_f32_16x16x32_f16(
                false, afrag, false, bfrag, (short)0, a3, false, false);
        }

        // Rows 0..kHCh[h]-1 live in VGPRs 0..2 of lanes 0-15 (hi==0 half).
        if (hi == 0) {
#pragma unroll
            for (int r = 0; r < kHCh[h]; ++r) {
                float v = a3[r] + b2E[h * 16 + r];
                out[((size_t)b * 12 + kChOff[h] + r) * NPTS + n0 + col] = v;
            }
        }
    }
}

// ---------------------------------------------------------------------------
extern "C" void kernel_launch(void* const* d_in, const int* in_sizes, int n_in,
                              void* d_out, int out_size, void* d_ws, size_t ws_size,
                              hipStream_t stream) {
    const float* ct  = (const float*)d_in[0];
    const float* sw  = (const float*)d_in[1];
    const float* sb  = (const float*)d_in[2];
    const float* sg  = (const float*)d_in[3];
    const float* sbe = (const float*)d_in[4];
    const float* sm  = (const float*)d_in[5];
    const float* sv  = (const float*)d_in[6];
    const float* w1  = (const float*)d_in[7];
    const float* b1  = (const float*)d_in[8];
    const float* g1  = (const float*)d_in[9];
    const float* be1 = (const float*)d_in[10];
    const float* m1  = (const float*)d_in[11];
    const float* v1  = (const float*)d_in[12];
    const float* w2  = (const float*)d_in[13];
    const float* b2  = (const float*)d_in[14];

    _Float16* wts  = (_Float16*)d_ws;
    float*    bias = (float*)((char*)d_ws + BIAS_OFF_BYTES);
    float*    bsE  = bias;
    float*    b1E  = bias + 64;
    float*    b2E  = bias + 64 + HN * CS;

    prep_kernel<<<dim3((HN * CS * CS + 255) / 256), dim3(256), 0, stream>>>(
        sw, sb, sg, sbe, sm, sv, w1, b1, g1, be1, m1, v1, w2, b2,
        wts + WSE_H, wts + W1E_H, wts + W2E_H, bsE, b1E, b2E);

    // 8 batches * 128 column-tiles of 128; 8 waves/block, 16 cols/wave.
    fused_head_kernel<<<dim3(NB * 128), dim3(256), 0, stream>>>(
        ct, wts, bsE, b1E, b2E, (float*)d_out);
}
